// LSTMSimple_24592982737380
// MI455X (gfx1250) — matrix operand order repaired and checked by reading, not compiled, not run on hardware
//
#include <hip/hip_runtime.h>

#define TT 512   // timesteps
#define BB 128   // batch
#define HH 512   // hidden
#define FF 512   // features
#define NG 2048  // 4*H gate width

typedef __attribute__((ext_vector_type(16))) __bf16 v16bf;
typedef __attribute__((ext_vector_type(8)))  float  v8f;

union FragBF { v16bf v; uint4 q[2]; };

__device__ __forceinline__ unsigned short f2bf(float f) {
  unsigned u = __float_as_uint(f);
  unsigned r = u + 0x7FFFu + ((u >> 16) & 1u);   // round-to-nearest-even
  return (unsigned short)(r >> 16);
}
__device__ __forceinline__ float bf2f(unsigned short u) {
  return __uint_as_float(((unsigned)u) << 16);
}
__device__ __forceinline__ float sigmoid_f(float x) { return 1.0f / (1.0f + __expf(-x)); }
__device__ __forceinline__ float tanh_f(float x) {
  float ax = fabsf(x);
  float e  = __expf(-2.0f * ax);
  float t  = (1.0f - e) / (1.0f + e);
  return x < 0.0f ? -t : t;
}

// ---------------- elementwise converts ----------------
__global__ void k_cvt_bf16(const float* __restrict__ src, unsigned short* __restrict__ dst, int n) {
  int i = blockIdx.x * blockDim.x + threadIdx.x;
  if (i < n) dst[i] = f2bf(src[i]);
}

// src is K x N (row-major), dst is N x K (row-major) bf16
__global__ void k_transpose_bf16(const float* __restrict__ src, unsigned short* __restrict__ dst,
                                 int K, int N) {
  int i = blockIdx.x * blockDim.x + threadIdx.x;
  if (i >= K * N) return;
  int n = i / K, k = i - n * K;
  dst[i] = f2bf(src[(size_t)k * N + n]);
}

// ---------------- big parallel GEMM: Z = A @ B^T + bias ----------------
// A: [M=B*T][512] bf16 row-major.  BT: [2048][512] bf16 (pre-transposed weights).
// Output Zt stored t-major: Zt[(t*BB + b)*NG + col], m = b*TT + t.
// Block = 256 threads (8 waves); block tile 128 rows x 64 cols; wave tile 16x64.
__global__ void __launch_bounds__(256)
k_gemm_bias(const unsigned short* __restrict__ A, const unsigned short* __restrict__ BT,
            const float* __restrict__ bias, float* __restrict__ Zt) {
  const int wave = threadIdx.x >> 5;
  const int lane = threadIdx.x & 31;
  const int lm = lane & 15, hi = lane >> 4;
  const int m0 = blockIdx.x * 128 + wave * 16;
  const int n0 = blockIdx.y * 64;

  const unsigned short* arow = A + (size_t)(m0 + lm) * FF;
  v8f acc[4] = {};
#pragma unroll 4
  for (int k0 = 0; k0 < FF; k0 += 32) {
    FragBF a;
    a.q[0] = *(const uint4*)(arow + k0 + hi * 8);        // K = k0 + hi*8 .. +7
    a.q[1] = *(const uint4*)(arow + k0 + 16 + hi * 8);   // K = k0+16+hi*8 .. +7
#pragma unroll
    for (int j = 0; j < 4; ++j) {
      const unsigned short* brow = BT + (size_t)(n0 + j * 16 + lm) * FF + k0 + hi * 8;
      FragBF b;
      b.q[0] = *(const uint4*)(brow);
      b.q[1] = *(const uint4*)(brow + 16);
      acc[j] = __builtin_amdgcn_wmma_f32_16x16x32_bf16(false, a.v, false, b.v,
                                                       (short)0, acc[j], false, false);
    }
  }
#pragma unroll
  for (int j = 0; j < 4; ++j) {
    int col = n0 + j * 16 + lm;
    float bv = bias[col];
#pragma unroll
    for (int r = 0; r < 8; ++r) {
      int m = m0 + hi * 8 + r;
      int t = m & (TT - 1), b = m >> 9;                  // m = b*TT + t
      Zt[((size_t)t * BB + b) * NG + col] = acc[j][r] + bv;
    }
  }
}

// ---------------- persistent recurrent LSTM kernel ----------------
// Grid = 64 blocks (32 j-tiles x 2 row-halves), block = 128 threads (4 waves).
// Wave w owns row-tile m = half*4 + w (16 batch rows), columns j0..j0+15, all 4 gates.
// Cell state c stays in VGPRs (C/D layout) for all T steps. Wh slice staged in LDS.
__global__ void __launch_bounds__(128)
k_lstm(const float* __restrict__ Zt, const unsigned short* __restrict__ WhT,
       unsigned short* __restrict__ hEven, unsigned short* __restrict__ hOdd,
       unsigned short* __restrict__ H1out, unsigned* __restrict__ bar) {
  __shared__ unsigned short ldsw[4 * 16 * HH];  // 64 KB: [gate][n(16)][k(512)]

  const int jt = blockIdx.x >> 1;
  const int half = blockIdx.x & 1;
  const int j0 = jt << 4;
  const int wave = threadIdx.x >> 5;
  const int lane = threadIdx.x & 31;
  const int lm = lane & 15, hi = lane >> 4;
  const int rowbase = (half * 4 + wave) << 4;

  // stage this block's Wh slice into LDS (once)
  for (int e = threadIdx.x * 8; e < 4 * 16 * HH; e += blockDim.x * 8) {
    int g = e >> 13;            // /8192
    int rem = e & 8191;
    int n = rem >> 9;
    int k = rem & 511;
    *(uint4*)(ldsw + e) = *(const uint4*)(WhT + ((size_t)(g * HH + j0 + n)) * HH + k);
  }
  __syncthreads();

  v8f c = {};
  for (int t = 0; t < TT; ++t) {
    const unsigned short* hr = (t & 1) ? hOdd : hEven;   // read h_{t-1}
    unsigned short* hw = (t & 1) ? hEven : hOdd;         // write h_t

    v8f acc[4] = {};
    const unsigned short* arow = hr + (size_t)(rowbase + lm) * HH;
    const unsigned short* bbase = ldsw + lm * HH;
#pragma unroll 2
    for (int k0 = 0; k0 < HH; k0 += 32) {
      FragBF a;
      a.q[0] = *(const uint4*)(arow + k0 + hi * 8);
      a.q[1] = *(const uint4*)(arow + k0 + 16 + hi * 8);
      const unsigned short* bp = bbase + k0 + hi * 8;
#pragma unroll
      for (int g = 0; g < 4; ++g) {
        FragBF b;
        b.q[0] = *(const uint4*)(bp + g * 8192);         // ds_load_b128
        b.q[1] = *(const uint4*)(bp + g * 8192 + 16);
        acc[g] = __builtin_amdgcn_wmma_f32_16x16x32_bf16(false, a.v, false, b.v,
                                                         (short)0, acc[g], false, false);
      }
    }

    const float* zrow = Zt + (size_t)t * (BB * NG);
#pragma unroll
    for (int r = 0; r < 8; ++r) {
      int brow = rowbase + hi * 8 + r;
      const float* zp = zrow + (size_t)brow * NG + j0 + lm;
      float iv = sigmoid_f(acc[0][r] + zp[0]);
      float fv = sigmoid_f(acc[1][r] + zp[HH]);
      float gv = tanh_f(acc[2][r] + zp[2 * HH]);
      float ov = sigmoid_f(acc[3][r] + zp[3 * HH]);
      float cn = fv * c[r] + iv * gv;
      c[r] = cn;
      float h = ov * tanh_f(cn);
      unsigned short hb = f2bf(h);
      hw[(size_t)brow * HH + j0 + lm] = hb;
      if (H1out) H1out[((size_t)brow * TT + t) * HH + j0 + lm] = hb;
    }

    // device-wide barrier (monotonic counter; reset each launch by memset)
    __syncthreads();
    __threadfence();
    if (threadIdx.x == 0) {
      __hip_atomic_fetch_add(bar, 1u, __ATOMIC_RELEASE, __HIP_MEMORY_SCOPE_AGENT);
      const unsigned target = (unsigned)gridDim.x * (unsigned)(t + 1);
      while (__hip_atomic_load(bar, __ATOMIC_ACQUIRE, __HIP_MEMORY_SCOPE_AGENT) < target)
        __builtin_amdgcn_s_sleep(2);
    }
    __syncthreads();
    __threadfence();
  }
}

// ---------------- batchnorm (two-stage, deterministic) ----------------
__global__ void k_bn_stage1(const unsigned short* __restrict__ X, int nrows,
                            float* __restrict__ psum, float* __restrict__ psq) {
  int j = threadIdx.x;                       // 512 channels
  int nb = gridDim.x;
  int chunk = (nrows + nb - 1) / nb;
  int r0 = blockIdx.x * chunk;
  int r1 = min(r0 + chunk, nrows);
  float s = 0.f, s2 = 0.f;
  for (int i = r0; i < r1; ++i) {
    float v = bf2f(X[(size_t)i * HH + j]);
    s += v; s2 += v * v;
  }
  psum[blockIdx.x * HH + j] = s;
  psq[blockIdx.x * HH + j] = s2;
}

__global__ void k_bn_stage2(const float* __restrict__ psum, const float* __restrict__ psq,
                            int nb, float nrows, const float* __restrict__ scale,
                            const float* __restrict__ bias,
                            float* __restrict__ aOut, float* __restrict__ bOut) {
  int j = threadIdx.x;
  float s = 0.f, s2 = 0.f;
  for (int k = 0; k < nb; ++k) { s += psum[k * HH + j]; s2 += psq[k * HH + j]; }
  float mean = s / nrows;
  float var = s2 / nrows - mean * mean;
  float a = scale[j] * rsqrtf(var + 1e-5f);
  aOut[j] = a;
  bOut[j] = bias[j] - mean * a;
}

__global__ void k_bn_apply(unsigned short* __restrict__ X, const float* __restrict__ a,
                           const float* __restrict__ b, size_t n) {
  size_t i = ((size_t)blockIdx.x * blockDim.x + threadIdx.x) * 8;
  if (i >= n) return;
  uint4 pk = *(uint4*)(X + i);
  unsigned short* e = (unsigned short*)&pk;
  int j0 = (int)(i & (HH - 1));
#pragma unroll
  for (int k = 0; k < 8; ++k) e[k] = f2bf(a[j0 + k] * bf2f(e[k]) + b[j0 + k]);
  *(uint4*)(X + i) = pk;
}

// ---------------- MLP head: out = tanh(hn @ Wd1 + bd1) @ Wd2 + bd2 ----------------
__global__ void __launch_bounds__(512)
k_head(const unsigned short* __restrict__ hT, const float* __restrict__ a2,
       const float* __restrict__ b2, const float* __restrict__ Wd1,
       const float* __restrict__ bd1, const float* __restrict__ Wd2,
       const float* __restrict__ bd2, float* __restrict__ out) {
  int b = blockIdx.x;
  int j = threadIdx.x;
  __shared__ float hn[HH];
  __shared__ float y1[16];
  hn[j] = a2[j] * bf2f(hT[(size_t)b * HH + j]) + b2[j];
  __syncthreads();
  if (j < 16) {
    float s = bd1[j];
    for (int k = 0; k < HH; ++k) s += hn[k] * Wd1[k * 16 + j];
    y1[j] = tanh_f(s);
  }
  __syncthreads();
  if (j == 0) {
    float s = bd2[0];
    for (int k = 0; k < 16; ++k) s += y1[k] * Wd2[k];
    out[b] = s;
  }
}

// ---------------- host orchestration ----------------
extern "C" void kernel_launch(void* const* d_in, const int* in_sizes, int n_in,
                              void* d_out, int out_size, void* d_ws, size_t ws_size,
                              hipStream_t stream) {
  (void)in_sizes; (void)n_in; (void)out_size; (void)ws_size;
  const float* x    = (const float*)d_in[0];
  const float* Wi1  = (const float*)d_in[1];
  const float* Wh1  = (const float*)d_in[2];
  const float* b1   = (const float*)d_in[3];
  const float* Wi2  = (const float*)d_in[4];
  const float* Wh2  = (const float*)d_in[5];
  const float* b2   = (const float*)d_in[6];
  const float* bn1s = (const float*)d_in[7];
  const float* bn1b = (const float*)d_in[8];
  const float* bn2s = (const float*)d_in[9];
  const float* bn2b = (const float*)d_in[10];
  const float* Wd1  = (const float*)d_in[11];
  const float* bd1  = (const float*)d_in[12];
  const float* Wd2  = (const float*)d_in[13];
  const float* bd2  = (const float*)d_in[14];
  float* out = (float*)d_out;

  char* w = (char*)d_ws;
  size_t off = 0;
  auto alloc = [&](size_t bytes) -> char* {
    char* p = w + off;
    off += (bytes + 255) & ~(size_t)255;
    return p;
  };
  const int M = BB * TT;  // 65536
  unsigned short* Xbf  = (unsigned short*)alloc((size_t)M * FF * 2);     // 64 MB
  unsigned short* WiT1 = (unsigned short*)alloc((size_t)NG * FF * 2);    // 2 MB
  unsigned short* WhT1 = (unsigned short*)alloc((size_t)NG * HH * 2);
  unsigned short* WiT2 = (unsigned short*)alloc((size_t)NG * HH * 2);
  unsigned short* WhT2 = (unsigned short*)alloc((size_t)NG * HH * 2);
  float*          Zt   = (float*)alloc((size_t)TT * BB * NG * 4);        // 512 MB
  unsigned short* H1   = (unsigned short*)alloc((size_t)M * HH * 2);     // 64 MB
  unsigned short* hA   = (unsigned short*)alloc((size_t)BB * HH * 2);
  unsigned short* hB   = (unsigned short*)alloc((size_t)BB * HH * 2);
  float* psum = (float*)alloc(64 * HH * 4);
  float* psq  = (float*)alloc(64 * HH * 4);
  float* a1   = (float*)alloc(HH * 4);
  float* bb1  = (float*)alloc(HH * 4);
  float* a2   = (float*)alloc(HH * 4);
  float* bb2  = (float*)alloc(HH * 4);
  unsigned* bars = (unsigned*)alloc(512);

  // convert activations + pre-transpose weights to bf16
  k_cvt_bf16<<<(M * FF + 255) / 256, 256, 0, stream>>>(x, Xbf, M * FF);
  k_transpose_bf16<<<(FF * NG + 255) / 256, 256, 0, stream>>>(Wi1, WiT1, FF, NG);
  k_transpose_bf16<<<(HH * NG + 255) / 256, 256, 0, stream>>>(Wh1, WhT1, HH, NG);
  k_transpose_bf16<<<(HH * NG + 255) / 256, 256, 0, stream>>>(Wi2, WiT2, HH, NG);
  k_transpose_bf16<<<(HH * NG + 255) / 256, 256, 0, stream>>>(Wh2, WhT2, HH, NG);

  dim3 ggrid(M / 128, NG / 64);

  // ---- layer 1 ----
  hipMemsetAsync(hA, 0, (size_t)BB * HH * 2, stream);
  hipMemsetAsync(hB, 0, (size_t)BB * HH * 2, stream);
  hipMemsetAsync(bars, 0, 512, stream);
  k_gemm_bias<<<ggrid, 256, 0, stream>>>(Xbf, WiT1, b1, Zt);
  k_lstm<<<64, 128, 0, stream>>>(Zt, WhT1, hA, hB, H1, bars);

  // ---- batchnorm over (B,T) ----
  k_bn_stage1<<<64, 512, 0, stream>>>(H1, M, psum, psq);
  k_bn_stage2<<<1, 512, 0, stream>>>(psum, psq, 64, (float)M, bn1s, bn1b, a1, bb1);
  k_bn_apply<<<(int)(((size_t)M * HH / 8 + 255) / 256), 256, 0, stream>>>(H1, a1, bb1, (size_t)M * HH);

  // ---- layer 2 ----
  hipMemsetAsync(hA, 0, (size_t)BB * HH * 2, stream);
  hipMemsetAsync(hB, 0, (size_t)BB * HH * 2, stream);
  k_gemm_bias<<<ggrid, 256, 0, stream>>>(H1, WiT2, b2, Zt);
  k_lstm<<<64, 128, 0, stream>>>(Zt, WhT2, hA, hB, nullptr, bars + 64);
  // t=511 (odd) writes hA -> final h_T lives in hA

  // ---- batchnorm over B (last step) + head ----
  k_bn_stage1<<<1, 512, 0, stream>>>(hA, BB, psum, psq);
  k_bn_stage2<<<1, 512, 0, stream>>>(psum, psq, 1, (float)BB, bn2s, bn2b, a2, bb2);
  k_head<<<BB, 512, 0, stream>>>(hA, a2, bb2, Wd1, bd1, Wd2, bd2, out);
}
